// Dec_9285719294075
// MI455X (gfx1250) — matrix operand order repaired and checked by reading, not compiled, not run on hardware
//
#include <hip/hip_runtime.h>

// ---------------- problem constants ----------------
#define BB    32      // batch
#define SS    60      // source length
#define TT    40      // target length
#define VV    50000   // zh vocab
#define HH    512     // ZH_HIDDEN
#define DD    256     // ZH_DIMS (emb)
#define RENC  512     // RNN_ENC
#define CENC  256     // CNN_ENC
#define ATTD  256     // ATT
#define KIN   1536    // 1024 (atten|cnn|emb) + 512 (hx)  -> fused gates K

typedef __attribute__((ext_vector_type(16))) __bf16 v16bf;
typedef __attribute__((ext_vector_type(8)))  float  v8f;

union FragBF { v16bf v; uint4 q[2]; };

static __device__ __forceinline__ unsigned short f2bf(float f) {
    unsigned int u = __float_as_uint(f);
    unsigned int r = 0x7FFFu + ((u >> 16) & 1u);   // round-to-nearest-even
    return (unsigned short)((u + r) >> 16);
}
static __device__ __forceinline__ float sigmoidf(float x) {
    return 1.0f / (1.0f + __expf(-x));
}

// ---------------- one-time setup kernels ----------------

// W_fc f32 [V,H] -> bf16 (stays resident in 192MB L2 across all 40 steps)
__global__ void k_cvt_wfc(const float* __restrict__ w, unsigned short* __restrict__ o, int n) {
    for (int i = blockIdx.x * 256 + threadIdx.x; i < n; i += gridDim.x * 256)
        o[i] = f2bf(w[i]);
}

// fuse W_ih [2048,1024] and W_hh [2048,512] into Wg bf16 [2048,1536]
__global__ void k_cvt_wg(const float* __restrict__ W_ih, const float* __restrict__ W_hh,
                         unsigned short* __restrict__ Wg) {
    int idx = blockIdx.x * 256 + threadIdx.x;
    if (idx >= 4 * HH * KIN) return;
    int row = idx / KIN, col = idx % KIN;
    float v = (col < 1024) ? W_ih[row * 1024 + col] : W_hh[row * HH + (col - 1024)];
    Wg[idx] = f2bf(v);
}

// Uh[b,s,a] = b_uh[a] + dot(rnn_enc[b,s,:], W_uh[a,:])   (one-time, 0.5 GFLOP)
__global__ void k_uh(const float* __restrict__ rnn_enc, const float* __restrict__ W_uh,
                     const float* __restrict__ b_uh, float* __restrict__ Uh) {
    int idx = blockIdx.x * 256 + threadIdx.x;
    if (idx >= BB * SS * ATTD) return;
    int a = idx & (ATTD - 1);
    int bs = idx >> 8;
    float acc = b_uh[a];
    const float* r = rnn_enc + (size_t)bs * RENC;
    const float* w = W_uh + (size_t)a * RENC;
    for (int k = 0; k < RENC; ++k) acc += r[k] * w[k];
    Uh[idx] = acc;
}

// zero hx, cx, and the hx-region of the fused input buffer
__global__ void k_init(float* __restrict__ hx, float* __restrict__ cx,
                       unsigned short* __restrict__ inp_ext) {
    int i = blockIdx.x * 256 + threadIdx.x;      // 0 .. B*H-1
    hx[i] = 0.0f; cx[i] = 0.0f;
    int b = i >> 9, h = i & (HH - 1);
    inp_ext[b * KIN + 1024 + h] = 0;             // bf16 zero
}

// ---------------- per-timestep kernels ----------------

// Bahdanau attention + build fused input row [atten | cnn | emb] (bf16)
__global__ void k_attn(const float* __restrict__ rnn_enc, const float* __restrict__ cnn_enc,
                       const int* __restrict__ gtruths, const float* __restrict__ emb,
                       const float* __restrict__ W_ws, const float* __restrict__ b_ws,
                       const float* __restrict__ W_v, const float* __restrict__ b_v,
                       const float* __restrict__ Uh, const float* __restrict__ hx,
                       unsigned short* __restrict__ inp_ext, int t) {
    __shared__ float sWs[ATTD];
    __shared__ float sScore[SS];
    __shared__ float sAlpha[SS];
    int b = blockIdx.x, tid = threadIdx.x;

    // ws = hx @ W_ws^T + b_ws  (one output per thread)
    {
        float acc = b_ws[tid];
        const float* h = hx + b * HH;
        const float* w = W_ws + tid * HH;
        for (int k = 0; k < HH; ++k) acc += h[k] * w[k];
        sWs[tid] = acc;
    }
    __syncthreads();

    // score[s] = tanh(Uh + ws) . W_v + b_v
    if (tid < SS) {
        float acc = b_v[0];
        const float* u = Uh + (size_t)(b * SS + tid) * ATTD;
        for (int a = 0; a < ATTD; ++a) acc += tanhf(u[a] + sWs[a]) * W_v[a];
        sScore[tid] = acc;
    }
    __syncthreads();

    // softmax over 60 source positions (serial on one lane; negligible)
    if (tid == 0) {
        float mx = sScore[0];
        for (int s = 1; s < SS; ++s) mx = fmaxf(mx, sScore[s]);
        float sum = 0.0f;
        for (int s = 0; s < SS; ++s) { float e = __expf(sScore[s] - mx); sAlpha[s] = e; sum += e; }
        float inv = 1.0f / sum;
        for (int s = 0; s < SS; ++s) sAlpha[s] *= inv;
    }
    __syncthreads();

    // atten = alpha @ rnn_enc  -> inp_ext[0:512]
    for (int e = tid; e < RENC; e += 256) {
        float acc = 0.0f;
        const float* re = rnn_enc + (size_t)b * SS * RENC + e;
        for (int s = 0; s < SS; ++s) acc += sAlpha[s] * re[s * RENC];
        inp_ext[b * KIN + e] = f2bf(acc);
    }
    // cnn_enc -> inp_ext[512:768]
    if (tid < CENC) inp_ext[b * KIN + RENC + tid] = f2bf(cnn_enc[b * CENC + tid]);
    // teacher-forced embedding -> inp_ext[768:1024]
    int tok = gtruths[b * TT + t];
    if (tid < DD) inp_ext[b * KIN + RENC + CENC + tid] = f2bf(emb[(size_t)tok * DD + tid]);
}

// gates = inp_ext @ Wg^T + (b_ih + b_hh)   [32 x 2048], K = 1536  -- WMMA bf16
// 16 blocks x 8 waves; each wave: 1 N-tile x 2 M-tiles; A staged in LDS (2 phases x 48KB)
__global__ void k_gates(const unsigned short* __restrict__ inp_ext,
                        const unsigned short* __restrict__ Wg,
                        const float* __restrict__ b_ih, const float* __restrict__ b_hh,
                        float* __restrict__ gates) {
    __shared__ __align__(16) unsigned int As[32 * 384];   // 32 rows x 768 bf16 (half of K)
    int wid = threadIdx.x >> 5, lane = threadIdx.x & 31;
    int ntile = blockIdx.x * 8 + wid;                     // 0..127
    int half = lane >> 4, l15 = lane & 15;
    v8f acc0 = {}, acc1 = {};
    const unsigned int* src = (const unsigned int*)inp_ext;   // 768 dwords per row

    for (int ph = 0; ph < 2; ++ph) {
        __syncthreads();
        for (int i = threadIdx.x; i < 32 * 384; i += 256) {
            int r = i / 384, c = i % 384;
            As[i] = src[r * 768 + ph * 384 + c];
        }
        __syncthreads();
        const uint4* wrow = (const uint4*)(Wg + (size_t)(ntile * 16 + l15) * KIN + ph * 768);
        for (int kk = 0; kk < 24; ++kk) {
            int k0 = kk * 32;
            FragBF fb;                                   // B: ISA 16-bit layout (same K order as A)
            int kb = k0 + half * 8;
            fb.q[0] = wrow[kb >> 3];
            fb.q[1] = wrow[(kb + 16) >> 3];
            int ka = k0 + half * 8;                      // A: ISA 16-bit A layout
            FragBF fa0, fa1;
            fa0.q[0] = *(const uint4*)&As[l15 * 384 + (ka >> 1)];
            fa0.q[1] = *(const uint4*)&As[l15 * 384 + ((ka + 16) >> 1)];
            fa1.q[0] = *(const uint4*)&As[(16 + l15) * 384 + (ka >> 1)];
            fa1.q[1] = *(const uint4*)&As[(16 + l15) * 384 + ((ka + 16) >> 1)];
            acc0 = __builtin_amdgcn_wmma_f32_16x16x32_bf16(false, fa0.v, false, fb.v,
                                                           (short)0, acc0, false, false);
            acc1 = __builtin_amdgcn_wmma_f32_16x16x32_bf16(false, fa1.v, false, fb.v,
                                                           (short)0, acc1, false, false);
        }
    }
    int n = ntile * 16 + l15;
    float bias = b_ih[n] + b_hh[n];
#pragma unroll
    for (int v = 0; v < 8; ++v) {                        // D: VGPR v -> row v (+8 upper half)
        int m = v + half * 8;
        gates[(m)      * 4 * HH + n] = acc0[v] + bias;
        gates[(16 + m) * 4 * HH + n] = acc1[v] + bias;
    }
}

// LSTM pointwise: c' = sig(f)*c + sig(i)*tanh(g); h' = sig(o)*tanh(c')
// also writes h' (bf16) into the hx slot of inp_ext for next step + FC A operand
__global__ void k_lstm(const float* __restrict__ gates, float* __restrict__ hx,
                       float* __restrict__ cx, unsigned short* __restrict__ inp_ext) {
    int i = blockIdx.x * 256 + threadIdx.x;              // 0 .. B*H-1
    int b = i >> 9, h = i & (HH - 1);
    const float* g = gates + (size_t)b * 4 * HH;
    float gi = g[h], gf = g[HH + h], gg = g[2 * HH + h], go = g[3 * HH + h];
    float c = sigmoidf(gf) * cx[i] + sigmoidf(gi) * tanhf(gg);
    float hn = sigmoidf(go) * tanhf(c);
    cx[i] = c; hx[i] = hn;
    inp_ext[b * KIN + 1024 + h] = f2bf(hn);
}

// logits = hx2 @ W_fc^T + b_fc   [32 x 50000], K = 512  -- WMMA bf16
// A (32x512 bf16 = 32KB) staged in LDS; each wave: 1 N-tile x 2 M-tiles x 16 K-steps
__global__ void k_fc(const unsigned short* __restrict__ inp_ext,
                     const unsigned short* __restrict__ Wfc,
                     const float* __restrict__ bfc, float* __restrict__ out) {
    __shared__ __align__(16) unsigned int As[32 * 256];  // 32 rows x 512 bf16
    const unsigned int* src = (const unsigned int*)inp_ext;  // hx at dword cols 512..767
    for (int i = threadIdx.x; i < 32 * 256; i += 256) {
        int r = i >> 8, c = i & 255;
        As[i] = src[r * 768 + 512 + c];
    }
    __syncthreads();
    int wid = threadIdx.x >> 5, lane = threadIdx.x & 31;
    int ntile = blockIdx.x * 8 + wid;
    if (ntile >= VV / 16) return;
    int half = lane >> 4, l15 = lane & 15;
    v8f acc0 = {}, acc1 = {};
    int nrow = ntile * 16 + l15;
    const uint4* wrow = (const uint4*)(Wfc + (size_t)nrow * HH);   // 64 uint4 per row
#pragma unroll 4
    for (int kk = 0; kk < 16; ++kk) {
        int k0 = kk * 32;
        FragBF fb;
        int kb = k0 + half * 8;
        fb.q[0] = wrow[kb >> 3];
        fb.q[1] = wrow[(kb + 16) >> 3];
        if (kk + 1 < 16)
            __builtin_prefetch((const void*)(wrow + ((k0 + 32) >> 3)), 0, 0);
        int ka = k0 + half * 8;
        FragBF fa0, fa1;
        fa0.q[0] = *(const uint4*)&As[l15 * 256 + (ka >> 1)];
        fa0.q[1] = *(const uint4*)&As[l15 * 256 + ((ka + 16) >> 1)];
        fa1.q[0] = *(const uint4*)&As[(16 + l15) * 256 + (ka >> 1)];
        fa1.q[1] = *(const uint4*)&As[(16 + l15) * 256 + ((ka + 16) >> 1)];
        acc0 = __builtin_amdgcn_wmma_f32_16x16x32_bf16(false, fa0.v, false, fb.v,
                                                       (short)0, acc0, false, false);
        acc1 = __builtin_amdgcn_wmma_f32_16x16x32_bf16(false, fa1.v, false, fb.v,
                                                       (short)0, acc1, false, false);
    }
    int ncol = ntile * 16 + l15;
    float bias = bfc[ncol];
#pragma unroll
    for (int v = 0; v < 8; ++v) {
        int m = v + half * 8;
        out[(size_t)(m)      * VV + ncol] = acc0[v] + bias;
        out[(size_t)(16 + m) * VV + ncol] = acc1[v] + bias;
    }
}

// argmax over V per (t,b); first-occurrence tie-break like jnp.argmax
__global__ void k_argmax(const float* __restrict__ logits, float* __restrict__ preds, int t) {
    __shared__ float smax[256];
    __shared__ int   sidx[256];
    int b = blockIdx.x;
    const float* row = logits + (size_t)b * VV;
    float best = -3.4e38f; int bi = 0;
    for (int j = threadIdx.x; j < VV; j += 256) {
        float v = row[j];
        if (v > best) { best = v; bi = j; }
    }
    smax[threadIdx.x] = best; sidx[threadIdx.x] = bi;
    __syncthreads();
    for (int s = 128; s > 0; s >>= 1) {
        if (threadIdx.x < s) {
            if (smax[threadIdx.x + s] > smax[threadIdx.x] ||
                (smax[threadIdx.x + s] == smax[threadIdx.x] &&
                 sidx[threadIdx.x + s] < sidx[threadIdx.x])) {
                smax[threadIdx.x] = smax[threadIdx.x + s];
                sidx[threadIdx.x] = sidx[threadIdx.x + s];
            }
        }
        __syncthreads();
    }
    if (threadIdx.x == 0) preds[b * TT + t] = (float)sidx[0];
}

// ---------------- host launcher ----------------
extern "C" void kernel_launch(void* const* d_in, const int* in_sizes, int n_in,
                              void* d_out, int out_size, void* d_ws, size_t ws_size,
                              hipStream_t stream) {
    const float* rnn_enc = (const float*)d_in[0];
    const float* cnn_enc = (const float*)d_in[1];
    const int*   gtruths = (const int*)d_in[2];
    const float* emb     = (const float*)d_in[3];
    const float* W_ih    = (const float*)d_in[4];
    const float* b_ih    = (const float*)d_in[5];
    const float* W_hh    = (const float*)d_in[6];
    const float* b_hh    = (const float*)d_in[7];
    const float* W_ws    = (const float*)d_in[8];
    const float* b_ws    = (const float*)d_in[9];
    const float* W_uh    = (const float*)d_in[10];
    const float* b_uh    = (const float*)d_in[11];
    const float* W_v     = (const float*)d_in[12];
    const float* b_v     = (const float*)d_in[13];
    const float* W_fc    = (const float*)d_in[14];
    const float* b_fc    = (const float*)d_in[15];
    // d_in[16]=ssprob(=1), d_in[17]=is_train(=1): pure teacher forcing, unused.

    char* ws = (char*)d_ws;
    unsigned short* Wfc_bf = (unsigned short*)ws;  ws += (size_t)VV * HH * 2;       // 51.2 MB
    unsigned short* Wg_bf  = (unsigned short*)ws;  ws += (size_t)4 * HH * KIN * 2;  // 6.3 MB
    float* Uh              = (float*)ws;           ws += (size_t)BB * SS * ATTD * 4;
    unsigned short* inp_ext= (unsigned short*)ws;  ws += (size_t)BB * KIN * 2;
    float* hx              = (float*)ws;           ws += (size_t)BB * HH * 4;
    float* cx              = (float*)ws;           ws += (size_t)BB * HH * 4;
    float* gates           = (float*)ws;           ws += (size_t)BB * 4 * HH * 4;

    float* out   = (float*)d_out;
    float* preds = out + (size_t)TT * BB * VV;

    k_cvt_wfc<<<4096, 256, 0, stream>>>(W_fc, Wfc_bf, VV * HH);
    k_cvt_wg<<<(4 * HH * KIN + 255) / 256, 256, 0, stream>>>(W_ih, W_hh, Wg_bf);
    k_uh<<<(BB * SS * ATTD + 255) / 256, 256, 0, stream>>>(rnn_enc, W_uh, b_uh, Uh);
    k_init<<<(BB * HH + 255) / 256, 256, 0, stream>>>(hx, cx, inp_ext);

    for (int t = 0; t < TT; ++t) {
        k_attn<<<BB, 256, 0, stream>>>(rnn_enc, cnn_enc, gtruths, emb, W_ws, b_ws,
                                       W_v, b_v, Uh, hx, inp_ext, t);
        k_gates<<<16, 256, 0, stream>>>(inp_ext, Wg_bf, b_ih, b_hh, gates);
        k_lstm<<<(BB * HH + 255) / 256, 256, 0, stream>>>(gates, hx, cx, inp_ext);
        float* logits_t = out + (size_t)t * BB * VV;
        k_fc<<<(VV / 16 + 7) / 8, 256, 0, stream>>>(inp_ext, Wfc_bf, b_fc, logits_t);
        k_argmax<<<BB, 256, 0, stream>>>(logits_t, preds, t);
    }
}
